// SelfAttention_36558761623849
// MI455X (gfx1250) — compile-verified, matrix operand order checked by reading, not run on hardware
//
#include <hip/hip_runtime.h>
#include <math.h>

// Problem constants (B=4, N=4096, D=512, H=8, DH=64, M=128)
#define R_TOT 16384   // B*N rows

typedef __attribute__((ext_vector_type(16))) __bf16 v16bf;
typedef __attribute__((ext_vector_type(8)))  float  v8f;

union Frag { v16bf v; unsigned u[8]; };

__device__ __forceinline__ unsigned short f2bf(float f) {
  unsigned u = __builtin_bit_cast(unsigned, f);
  u += 0x7fffu + ((u >> 16) & 1u);           // round-to-nearest-even
  return (unsigned short)(u >> 16);
}
__device__ __forceinline__ unsigned pack2bf(float lo, float hi) {
  return (unsigned)f2bf(lo) | ((unsigned)f2bf(hi) << 16);
}
// ISA 7.12.2: 16-bit A/B fragment pair index for vgpr-pair p on this lane.
// pair kp -> K = {2kp, 2kp+1}; kp = (p>=4 ? 8:0) + (lane>=16 ? 4:0) + (p&3)
__device__ __forceinline__ int kpair(int p, int lane) {
  return ((p & 4) << 1) + ((lane >> 4) << 2) + (p & 3);
}
__device__ __forceinline__ v8f wmma_bf16(v16bf a, v16bf b, v8f c) {
  return __builtin_amdgcn_wmma_f32_16x16x32_bf16(false, a, false, b, (short)0, c,
                                                 false, false);
}

// ---------------------------------------------------------------------------
// C[R,512] = A[R,512] x W[512,512], bf16 WMMA with f32 accumulate.
// 128 threads = 4 waves; block tile 64x64; wave tile 32x32 (2x2 WMMA); K step 32.
// ---------------------------------------------------------------------------
__global__ __launch_bounds__(128) void gemm512_kernel(
    const float* __restrict__ A, const float* __restrict__ W,
    float* __restrict__ C) {
  __shared__ unsigned short lA[64][36];   // row-major bf16, 72B row pad (bank-safe)
  __shared__ unsigned lB[16][64];         // pair-packed: [k/2][col] = (B[2k],B[2k+1])
  const int t = threadIdx.x, lane = t & 31, wave = t >> 5;
  const int r0 = blockIdx.x * 64, c0 = blockIdx.y * 64;
  const int wr = (wave >> 1) * 32, wc = (wave & 1) * 32;
  v8f acc[2][2] = {};
  for (int k0 = 0; k0 < 512; k0 += 32) {
    { // stage A: 64 rows x 32 k
      int ar = t >> 1, ac = (t & 1) * 16;
      const float* src = A + (size_t)(r0 + ar) * 512 + k0 + ac;
      unsigned* dst = (unsigned*)&lA[ar][ac];
#pragma unroll
      for (int j = 0; j < 8; ++j) dst[j] = pack2bf(src[2 * j], src[2 * j + 1]);
    }
    { // stage B: 32 k x 64 cols, pair-packed
      int p = t >> 3, cj = (t & 7) * 8;
      const float* w0 = W + (size_t)(k0 + 2 * p) * 512 + c0 + cj;
#pragma unroll
      for (int j = 0; j < 8; ++j) lB[p][cj + j] = pack2bf(w0[j], w0[512 + j]);
    }
    __syncthreads();
    Frag af[2], bf[2];
#pragma unroll
    for (int i = 0; i < 2; ++i) {
      int row = wr + i * 16 + (lane & 15);
#pragma unroll
      for (int p = 0; p < 8; ++p)
        af[i].u[p] = *(const unsigned*)&lA[row][kpair(p, lane) * 2];
    }
#pragma unroll
    for (int j = 0; j < 2; ++j) {
      int col = wc + j * 16 + (lane & 15);
#pragma unroll
      for (int p = 0; p < 8; ++p) bf[j].u[p] = lB[kpair(p, lane)][col];
    }
#pragma unroll
    for (int i = 0; i < 2; ++i)
#pragma unroll
      for (int j = 0; j < 2; ++j) acc[i][j] = wmma_bf16(af[i].v, bf[j].v, acc[i][j]);
    __syncthreads();
  }
#pragma unroll
  for (int i = 0; i < 2; ++i)
#pragma unroll
    for (int j = 0; j < 2; ++j) {
      int col = c0 + wc + j * 16 + (lane & 15);
      int rbase = r0 + wr + i * 16 + ((lane >> 4) << 3);
      float* out = C + (size_t)rbase * 512 + col;
#pragma unroll
      for (int r = 0; r < 8; ++r) out[(size_t)r * 512] = acc[i][j][r];
    }
}

// ---------------------------------------------------------------------------
// Fourier features: per (64-row tile, head h):
//   qp = scale*pp; kp = qp + off*scale*sp; dd = (dn*x) @ P^T (WMMA);
//   q'/k' = [ratio*sin(dd), ratio*cos(dd)]; norms = ||sp||/N.
// ---------------------------------------------------------------------------
__global__ __launch_bounds__(128) void fourier_kernel(
    const float* __restrict__ pp, const float* __restrict__ sp,
    const float* __restrict__ scale, const float* __restrict__ offs,
    const float* __restrict__ proj, float* __restrict__ qpO,
    float* __restrict__ kpO, float* __restrict__ norms) {
  __shared__ unsigned lP[32][128];           // P^T pair-packed: [d/2][m]
  __shared__ unsigned short lAq[64][68];     // 64 rows x 64 d (bf16, padded)
  __shared__ unsigned short lAk[64][68];
  __shared__ float lNrm[64][2];
  const int t = threadIdx.x, lane = t & 31, wave = t >> 5;
  const int h = blockIdx.y, r0 = blockIdx.x * 64;
  const float sc = scale[h], of = offs[h];
  const float dn = 0.3535533905932738f;      // 64^-0.25
  const float ratio = 0.08838834764831845f;  // 128^-0.5
  { // stage P^T: thread t owns proj row t (m = t); pairs contiguous in memory
    const float* pr = proj + (size_t)t * 64;
#pragma unroll
    for (int p = 0; p < 32; ++p) lP[p][t] = pack2bf(pr[2 * p], pr[2 * p + 1]);
  }
  { // stage q/k A tiles (+dn), norm partials
    int rr = t & 63, hh = t >> 6;
    const float* pps = pp + (size_t)(r0 + rr) * 512 + h * 64 + hh * 32;
    const float* sps = sp + (size_t)(r0 + rr) * 512 + h * 64 + hh * 32;
    float ss = 0.f;
#pragma unroll
    for (int j = 0; j < 32; j += 2) {
      float p0 = pps[j], p1 = pps[j + 1];
      float s0 = sps[j], s1 = sps[j + 1];
      float q0 = sc * p0, q1 = sc * p1;
      float k0 = q0 + of * sc * s0, k1 = q1 + of * sc * s1;
      *(unsigned*)&lAq[rr][hh * 32 + j] = pack2bf(dn * q0, dn * q1);
      *(unsigned*)&lAk[rr][hh * 32 + j] = pack2bf(dn * k0, dn * k1);
      ss += s0 * s0 + s1 * s1;
    }
    lNrm[rr][hh] = ss;
  }
  __syncthreads();
  if (t < 64)
    norms[(size_t)(r0 + t) * 8 + h] =
        sqrtf(lNrm[t][0] + lNrm[t][1]) * (1.0f / 4096.0f);
  const int rw = wave * 16;
  Frag aq[2], ak[2];
#pragma unroll
  for (int kk = 0; kk < 2; ++kk) {
    int row = rw + (lane & 15);
#pragma unroll
    for (int p = 0; p < 8; ++p) {
      int kp = kpair(p, lane);
      aq[kk].u[p] = *(const unsigned*)&lAq[row][kk * 32 + kp * 2];
      ak[kk].u[p] = *(const unsigned*)&lAk[row][kk * 32 + kp * 2];
    }
  }
#pragma unroll
  for (int s = 0; s < 2; ++s) {
    float* outp = s ? kpO : qpO;
    Frag* a = s ? ak : aq;
#pragma unroll
    for (int nt = 0; nt < 8; ++nt) {
      v8f acc = {};
#pragma unroll
      for (int kk = 0; kk < 2; ++kk) {
        Frag b;
#pragma unroll
        for (int p = 0; p < 8; ++p)
          b.u[p] = lP[kk * 16 + kpair(p, lane)][nt * 16 + (lane & 15)];
        acc = wmma_bf16(a[kk].v, b.v, acc);
      }
      int m = nt * 16 + (lane & 15);
#pragma unroll
      for (int r = 0; r < 8; ++r) {
        int grow = r0 + rw + r + ((lane >> 4) << 3);
        size_t base = ((size_t)grow * 8 + h) * 256;
        float dd = acc[r];
        outp[base + m]       = ratio * __sinf(dd);
        outp[base + 128 + m] = ratio * __cosf(dd);
      }
    }
  }
}

// ---------------------------------------------------------------------------
// kvs partial: per (b,h, K-chunk of 512 rows): kvs_c[256,64] = K'^T x V.
// 512 threads = 16 waves; wave owns a 16(m) x 64(d) strip (4 WMMA tiles).
// ---------------------------------------------------------------------------
__global__ __launch_bounds__(512) void kvs_partial_kernel(
    const float* __restrict__ kprime, const float* __restrict__ value,
    float* __restrict__ part) {
  __shared__ unsigned short lA[16][256][2];  // [l/2][m][l&1] transposed k'
  __shared__ unsigned short lB[16][64][2];   // [l/2][d][l&1] value
  const int t = threadIdx.x, lane = t & 31, wave = t >> 5;
  const int bh = blockIdx.x, chunk = blockIdx.y;
  const int b = bh >> 3, h = bh & 7;
  const int mw = wave * 16, lbase = chunk * 512;
  v8f acc[4] = {};
  for (int l0 = 0; l0 < 512; l0 += 32) {
    { int l = t >> 4, m0 = (t & 15) * 16;
      const float* src =
          kprime + ((size_t)(b * 4096 + lbase + l0 + l) * 8 + h) * 256 + m0;
#pragma unroll
      for (int j = 0; j < 16; ++j) lA[l >> 1][m0 + j][l & 1] = f2bf(src[j]);
    }
    { int l = t >> 4, d0 = (t & 15) * 4;
      const float* src =
          value + (size_t)(b * 4096 + lbase + l0 + l) * 512 + h * 64 + d0;
#pragma unroll
      for (int j = 0; j < 4; ++j) lB[l >> 1][d0 + j][l & 1] = f2bf(src[j]);
    }
    __syncthreads();
    Frag a;
#pragma unroll
    for (int p = 0; p < 8; ++p)
      a.u[p] = *(const unsigned*)&lA[kpair(p, lane)][mw + (lane & 15)][0];
#pragma unroll
    for (int nt = 0; nt < 4; ++nt) {
      Frag b2;
#pragma unroll
      for (int p = 0; p < 8; ++p)
        b2.u[p] = *(const unsigned*)&lB[kpair(p, lane)][nt * 16 + (lane & 15)][0];
      acc[nt] = wmma_bf16(a.v, b2.v, acc[nt]);
    }
    __syncthreads();
  }
#pragma unroll
  for (int nt = 0; nt < 4; ++nt)
#pragma unroll
    for (int r = 0; r < 8; ++r) {
      int m = mw + r + ((lane >> 4) << 3);
      part[(size_t)chunk * (32 * 256 * 64) +
           ((size_t)bh * 256 + m) * 64 + nt * 16 + (lane & 15)] = acc[nt][r];
    }
}

__global__ void kvs_reduce_kernel(const float* __restrict__ part,
                                  float* __restrict__ kvs) {
  size_t i = (size_t)blockIdx.x * 256 + threadIdx.x;
  float s = 0.f;
#pragma unroll
  for (int c = 0; c < 8; ++c) s += part[(size_t)c * (32 * 256 * 64) + i];
  kvs[i] = s;
}

// ---------------------------------------------------------------------------
// av: per (64-row tile, head): av[l,d] = norm[l] * (Q'[l,:256] @ kvs[256,64])
// ---------------------------------------------------------------------------
__global__ __launch_bounds__(128) void av_kernel(
    const float* __restrict__ qprime, const float* __restrict__ kvs,
    const float* __restrict__ norms, float* __restrict__ av) {
  __shared__ unsigned short lA[64][36];
  __shared__ unsigned short lB[16][64][2];
  __shared__ float lN[64];
  const int t = threadIdx.x, lane = t & 31, wave = t >> 5;
  const int r0 = blockIdx.x * 64, h = blockIdx.y;
  const int b = r0 >> 12;
  const float* kv = kvs + (size_t)(b * 8 + h) * 256 * 64;
  if (t < 64) lN[t] = norms[(size_t)(r0 + t) * 8 + h];
  const int wr = (wave >> 1) * 32, wc = (wave & 1) * 32;
  v8f acc[2][2] = {};
  for (int m0 = 0; m0 < 256; m0 += 32) {
    { int ar = t >> 1, ac = (t & 1) * 16;
      const float* src = qprime + ((size_t)(r0 + ar) * 8 + h) * 256 + m0 + ac;
      unsigned* dst = (unsigned*)&lA[ar][ac];
#pragma unroll
      for (int j = 0; j < 8; ++j) dst[j] = pack2bf(src[2 * j], src[2 * j + 1]);
    }
    { int m = t >> 2, d0 = (t & 3) * 16;
      const float* src = kv + (size_t)(m0 + m) * 64 + d0;
#pragma unroll
      for (int j = 0; j < 16; ++j) lB[m >> 1][d0 + j][m & 1] = f2bf(src[j]);
    }
    __syncthreads();
    Frag af[2], bf[2];
#pragma unroll
    for (int i = 0; i < 2; ++i) {
      int row = wr + i * 16 + (lane & 15);
#pragma unroll
      for (int p = 0; p < 8; ++p)
        af[i].u[p] = *(const unsigned*)&lA[row][kpair(p, lane) * 2];
    }
#pragma unroll
    for (int j = 0; j < 2; ++j) {
      int col = wc + j * 16 + (lane & 15);
#pragma unroll
      for (int p = 0; p < 8; ++p)
        bf[j].u[p] = *(const unsigned*)&lB[kpair(p, lane)][col][0];
    }
#pragma unroll
    for (int i = 0; i < 2; ++i)
#pragma unroll
      for (int j = 0; j < 2; ++j) acc[i][j] = wmma_bf16(af[i].v, bf[j].v, acc[i][j]);
    __syncthreads();
  }
#pragma unroll
  for (int i = 0; i < 2; ++i)
#pragma unroll
    for (int j = 0; j < 2; ++j)
#pragma unroll
      for (int r = 0; r < 8; ++r) {
        int rloc = wr + i * 16 + ((lane >> 4) << 3) + r;
        av[(size_t)(r0 + rloc) * 512 + h * 64 + wc + j * 16 + (lane & 15)] =
            lN[rloc] * acc[i][j][r];
      }
}

// ---------------------------------------------------------------------------
extern "C" void kernel_launch(void* const* d_in, const int* in_sizes, int n_in,
                              void* d_out, int out_size, void* d_ws,
                              size_t ws_size, hipStream_t stream) {
  (void)in_sizes; (void)n_in; (void)out_size; (void)ws_size;
  const float* src    = (const float*)d_in[0];  // source_input [B,N,D]
  const float* posf   = (const float*)d_in[1];  // pos_ft
  const float* slopes = (const float*)d_in[2];  // pos_ft_slopes
  const float* Wv     = (const float*)d_in[3];  // value_weight [512,512]
  const float* Wp     = (const float*)d_in[4];  // pos_ft_weight
  const float* scale  = (const float*)d_in[5];  // [8]
  const float* offs   = (const float*)d_in[6];  // [8]
  const float* Wo     = (const float*)d_in[7];  // output_weight [512,512]
  const float* proj   = (const float*)d_in[8];  // [128,64]

  float* out    = (float*)d_out;                       // [16384,512]
  float* qp_out = out + (size_t)R_TOT * 512;           // [16384,8,256]
  float* kp_out = qp_out + (size_t)R_TOT * 8 * 256;

  float* ws     = (float*)d_ws;
  float* ws_val = ws;                      // 8,388,608 f32
  float* ws_pp  = ws_val + 8388608;        // 8,388,608
  float* ws_sp  = ws_pp + 8388608;         // 8,388,608
  float* ws_nrm = ws_sp + 8388608;         //   131,072
  float* ws_kvs = ws_nrm + 131072;         //   524,288
  float* ws_part = ws_sp;                  // reuse sp (dead after fourier)
  float* ws_av   = ws_pp;                  // reuse pp (dead after fourier)

  dim3 blk128(128), g_gemm(R_TOT / 64, 8);
  gemm512_kernel<<<g_gemm, blk128, 0, stream>>>(src, Wv, ws_val);
  gemm512_kernel<<<g_gemm, blk128, 0, stream>>>(posf, Wp, ws_pp);
  gemm512_kernel<<<g_gemm, blk128, 0, stream>>>(slopes, Wp, ws_sp);
  fourier_kernel<<<dim3(R_TOT / 64, 8), blk128, 0, stream>>>(
      ws_pp, ws_sp, scale, offs, proj, qp_out, kp_out, ws_nrm);
  kvs_partial_kernel<<<dim3(32, 8), dim3(512), 0, stream>>>(kp_out, ws_val,
                                                            ws_part);
  kvs_reduce_kernel<<<dim3((32 * 256 * 64) / 256), dim3(256), 0, stream>>>(
      ws_part, ws_kvs);
  av_kernel<<<dim3(R_TOT / 64, 8), blk128, 0, stream>>>(qp_out, ws_kvs, ws_nrm,
                                                        ws_av);
  gemm512_kernel<<<g_gemm, blk128, 0, stream>>>(ws_av, Wo, out);
}
